// DecoderModel_33268816675399
// MI455X (gfx1250) — matrix operand order repaired and checked by reading, not compiled, not run on hardware
//
#include <hip/hip_runtime.h>
#include <hip/hip_bf16.h>

// MI455X / gfx1250, wave32. bf16 WMMA (16x16x32) with f32 accumulation.
typedef __bf16 bf16_t;
typedef __attribute__((ext_vector_type(16))) __bf16 v16bf;
typedef __attribute__((ext_vector_type(8)))  __bf16 v8bf;
typedef __attribute__((ext_vector_type(8)))  float  v8f;
typedef __attribute__((ext_vector_type(4)))  float  v4f;   // native vector for NT stores

#define DIM  128
#define PAD  144      // padded LDS row (elements); 288B rows keep 16B alignment, spread banks
#define TILE 128      // tokens per t-tile (per block iteration)
#define EPS  1e-5f

__device__ __forceinline__ v16bf pack16(v8bf lo, v8bf hi) {
  v16bf r;
#pragma unroll
  for (int i = 0; i < 8; ++i) { r[i] = lo[i]; r[i + 8] = hi[i]; }
  return r;
}

// Linear(128,128)+bias -> LayerNorm -> SiLU on this wave's 16-token tile.
// src/dst: [token][PAD] bf16 tiles in LDS. wl: [d][PAD] bf16 (row d = W[d][*]).
// cc: {bias[128], gamma[128], beta[128]} f32 in LDS.
__device__ __forceinline__ void layer16(const bf16_t* __restrict__ src,
                                        const bf16_t* __restrict__ wl,
                                        const float*  __restrict__ cc,
                                        bf16_t* __restrict__ dst,
                                        int tb, int lane) {
  const int m    = lane & 15;
  const int half = lane >> 4;

  // A fragments (16 tokens x 32 K per fragment). A layout: lane<16 holds
  // K {0..7,16..23}, lane>=16 holds K {8..15,24..31} of each 32-block.
  v16bf A[4];
  const bf16_t* arow = src + (size_t)(tb + m) * PAD;
#pragma unroll
  for (int k = 0; k < 4; ++k) {
    v8bf a0 = *(const v8bf*)(arow + k * 32 + half * 8);
    v8bf a1 = *(const v8bf*)(arow + k * 32 + 16 + half * 8);
    A[k] = pack16(a0, a1);
  }

  v8f acc[8];
#pragma unroll
  for (int j = 0; j < 8; ++j)
#pragma unroll
    for (int i = 0; i < 8; ++i) acc[j][i] = 0.f;

  // B layout: lane n%16 = column (out dim), lanes<16 hold K {0..7,16..23}, lanes>=16 K {8..15,24..31}.
  // Preload all 4 k-fragments of a j-block before its WMMA chain so the
  // scheduler can overlap ds_load latency with XDL work (partial dscnt waits).
#pragma unroll
  for (int j = 0; j < 8; ++j) {
    const bf16_t* brow = wl + (size_t)(j * 16 + m) * PAD + half * 8;
    v16bf B[4];
#pragma unroll
    for (int k = 0; k < 4; ++k)
      B[k] = pack16(*(const v8bf*)(brow + k * 32),
                    *(const v8bf*)(brow + k * 32 + 16));
#pragma unroll
    for (int k = 0; k < 4; ++k)
      acc[j] = __builtin_amdgcn_wmma_f32_16x16x32_bf16(
          false, A[k], false, B[k], (short)0, acc[j], false, false);
  }

  // + bias (per out-dim; C/D layout: out-dim = lane%16 within block j)
#pragma unroll
  for (int j = 0; j < 8; ++j) {
    float bj = cc[j * 16 + m];
#pragma unroll
    for (int r = 0; r < 8; ++r) acc[j][r] += bj;
  }

  // LayerNorm across the token's 128 outputs (rows of C live across 16 lanes),
  // then SiLU, then bf16 repack for the next layer's A operand.
#pragma unroll
  for (int r = 0; r < 8; ++r) {
    float s = 0.f, sq = 0.f;
#pragma unroll
    for (int j = 0; j < 8; ++j) { float v = acc[j][r]; s += v; sq += v * v; }
#pragma unroll
    for (int off = 1; off < 16; off <<= 1) {
      s  += __shfl_xor(s,  off, 32);
      sq += __shfl_xor(sq, off, 32);
    }
    float mean = s * (1.f / 128.f);
    float var  = sq * (1.f / 128.f) - mean * mean;
    float rs   = rsqrtf(var + EPS);
    bf16_t* drow = dst + (size_t)(tb + r + half * 8) * PAD;
#pragma unroll
    for (int j = 0; j < 8; ++j) {
      int d = j * 16 + m;
      float v = (acc[j][r] - mean) * rs * cc[DIM + d] + cc[2 * DIM + d];
      v = v * (1.f / (1.f + __expf(-v)));  // SiLU
      drow[d] = (bf16_t)v;
    }
  }
}

__global__ __launch_bounds__(256) void decoder_fused(
    const float* __restrict__ hidden,
    const float* __restrict__ W1, const float* __restrict__ b1,
    const float* __restrict__ g1, const float* __restrict__ be1,
    const float* __restrict__ W2, const float* __restrict__ b2,
    const float* __restrict__ g2, const float* __restrict__ be2,
    const float* __restrict__ Wm, const float* __restrict__ bm,
    const float* __restrict__ Ws, const float* __restrict__ bs,
    float* __restrict__ out) {
  __shared__ bf16_t Xs[TILE * PAD];     // activations (x, later h2)
  __shared__ bf16_t Hs[TILE * PAD];     // h1
  __shared__ bf16_t W1s[DIM * PAD];
  __shared__ bf16_t W2s[DIM * PAD];
  __shared__ bf16_t Whs[16 * PAD];      // [Wm; Ws]
  __shared__ float  c1[3 * DIM];
  __shared__ float  c2[3 * DIM];
  __shared__ float  bh[16];

  const int tid  = threadIdx.x;
  const int lane = tid & 31;
  const int wave = tid >> 5;
  const int m    = lane & 15;
  const int half = lane >> 4;
  const int tb   = wave * 16;

  // Stage weights/consts once per block (bf16; L2-resident so cheap to refetch)
  for (int idx = tid; idx < DIM * DIM; idx += 256) {
    int d = idx >> 7, c = idx & 127;
    W1s[d * PAD + c] = (bf16_t)W1[idx];
    W2s[d * PAD + c] = (bf16_t)W2[idx];
  }
  for (int idx = tid; idx < 16 * DIM; idx += 256) {
    int o = idx >> 7, c = idx & 127;
    Whs[o * PAD + c] = (bf16_t)((o < 8) ? Wm[o * DIM + c] : Ws[(o - 8) * DIM + c]);
  }
  if (tid < DIM) {
    c1[tid] = b1[tid]; c1[DIM + tid] = g1[tid]; c1[2 * DIM + tid] = be1[tid];
    c2[tid] = b2[tid]; c2[DIM + tid] = g2[tid]; c2[2 * DIM + tid] = be2[tid];
  }
  if (tid < 16) bh[tid] = (tid < 8) ? bm[tid] : bs[tid - 8];

  const int b = blockIdx.x >> 8;     // 8 batches
  const int n = blockIdx.x & 255;    // 256 spatial rows
  const size_t base_in = (size_t)b * (DIM * 65536) + (size_t)n * 256;
  float* out_mu = out;
  float* out_sg = out + (size_t)4194304;  // 8*8*256*256

  for (int t0 = 0; t0 < 256; t0 += TILE) {
    __syncthreads();  // Xs safe to overwrite; weights visible on 1st iter
    // hidden[b,c,n,t]: t contiguous -> coalesced; transpose into Xs[token][c].
    // Streamed once per block -> non-temporal so weights stay L2-resident.
    for (int idx = tid; idx < TILE * DIM; idx += 256) {
      int c = idx >> 7;
      int t = idx & 127;
      float v = __builtin_nontemporal_load(
          hidden + base_in + (size_t)c * 65536 + (t0 + t));
      Xs[t * PAD + c] = (bf16_t)v;
    }
    // Prefetch next t-tile (one global_prefetch_b8 per 128B line) so its HBM
    // latency overlaps this tile's WMMA work.
    if (t0 + TILE < 256) {
      for (int idx = tid; idx < 512; idx += 256) {
        int c = idx >> 2, chunk = idx & 3;
        __builtin_prefetch(
            hidden + base_in + (size_t)c * 65536 + (t0 + TILE) + chunk * 32,
            0, 1);
      }
    }
    __syncthreads();

    layer16(Xs, W1s, c1, Hs, tb, lane);  // h1 (same-wave LDS reuse, in-order)
    layer16(Hs, W2s, c2, Xs, tb, lane);  // h2 back into Xs (wave-private rows)

    // Heads: 16 tokens x 16 outputs (o<8 -> mu, o>=8 -> sigma)
    v8f acc;
#pragma unroll
    for (int i = 0; i < 8; ++i) acc[i] = 0.f;
    const bf16_t* arow = Xs + (size_t)(tb + m) * PAD;
    const bf16_t* brow = Whs + (size_t)m * PAD + half * 8;
    v16bf Bh[4];
#pragma unroll
    for (int k = 0; k < 4; ++k)
      Bh[k] = pack16(*(const v8bf*)(brow + k * 32),
                     *(const v8bf*)(brow + k * 32 + 16));
#pragma unroll
    for (int k = 0; k < 4; ++k) {
      v8bf a0 = *(const v8bf*)(arow + k * 32 + half * 8);
      v8bf a1 = *(const v8bf*)(arow + k * 32 + 16 + half * 8);
      acc = __builtin_amdgcn_wmma_f32_16x16x32_bf16(
          false, pack16(a0, a1), false, Bh[k], (short)0, acc, false, false);
    }
    float bias = bh[m];
    float ov[8];
#pragma unroll
    for (int r = 0; r < 8; ++r) {
      float v = acc[r] + bias;
      if (m >= 8) {  // sigma = 0.1 + 0.9*softplus(pre)
        float sp = (v > 20.f) ? v : log1pf(__expf(v));
        v = 0.1f + 0.9f * sp;
      }
      ov[r] = v;
    }
    // 8 consecutive t per lane -> two non-temporal 16B stores (write-once data)
    float* outp = ((m < 8) ? out_mu : out_sg)
                + (size_t)b * (8 * 65536) + (size_t)(m & 7) * 65536
                + (size_t)n * 256 + t0 + tb + half * 8;
    v4f lo = {ov[0], ov[1], ov[2], ov[3]};
    v4f hi = {ov[4], ov[5], ov[6], ov[7]};
    __builtin_nontemporal_store(lo, (v4f*)outp);
    __builtin_nontemporal_store(hi, (v4f*)(outp + 4));
  }
}

extern "C" void kernel_launch(void* const* d_in, const int* in_sizes, int n_in,
                              void* d_out, int out_size, void* d_ws, size_t ws_size,
                              hipStream_t stream) {
  (void)in_sizes; (void)n_in; (void)d_ws; (void)ws_size; (void)out_size;
  const float* hidden = (const float*)d_in[0];
  const float* W1  = (const float*)d_in[1];
  const float* b1  = (const float*)d_in[2];
  const float* g1  = (const float*)d_in[3];
  const float* be1 = (const float*)d_in[4];
  const float* W2  = (const float*)d_in[5];
  const float* b2  = (const float*)d_in[6];
  const float* g2  = (const float*)d_in[7];
  const float* be2 = (const float*)d_in[8];
  const float* Wm  = (const float*)d_in[9];
  const float* bm  = (const float*)d_in[10];
  const float* Ws  = (const float*)d_in[11];
  const float* bs  = (const float*)d_in[12];
  // 8 batches * 256 n-rows = 2048 blocks; each loops over two 128-token tiles
  decoder_fused<<<dim3(2048), dim3(256), 0, stream>>>(
      hidden, W1, b1, g1, be1, W2, b2, g2, be2, Wm, bm, Ws, bs, (float*)d_out);
}
